// SpatialAttention_36953898614883
// MI455X (gfx1250) — matrix operand order repaired and checked by reading, not compiled, not run on hardware
//
#include <hip/hip_runtime.h>
#include <stdint.h>

// Problem constants (from reference)
#define BB   4
#define CCH  384      // C
#define NN   4096     // W*H*Z
#define NH   8
#define DDIM 48       // C / NH
#define PKK  64       // PROJ
#define WQC  1536     // 4*C
#define YCOL 1152     // packed q|k|v columns (qkvv[2] slice skipped)

typedef __bf16 bf16x16 __attribute__((ext_vector_type(16)));
typedef float  f32x8   __attribute__((ext_vector_type(8)));
typedef unsigned int u32x4 __attribute__((ext_vector_type(4)));
typedef int          i32x4 __attribute__((ext_vector_type(4)));
typedef int          i32x8 __attribute__((ext_vector_type(8)));

union FragU { unsigned int u[8]; bf16x16 v; };

// TDM available on both toolchains; arity differs (ROCm7.2/clang-22: 5 args,
// amdgpu-toolchain/clang-23: 6 args with an extra i32x8 before cpol).
#if defined(__gfx1250__) && __has_builtin(__builtin_amdgcn_tensor_load_to_lds)
#define USE_TDM 1
#else
#define USE_TDM 0
#endif

__device__ __forceinline__ unsigned short f2bf(float f) {
  unsigned int u = __float_as_uint(f);
  u += 0x7fffu + ((u >> 16) & 1u);           // round-to-nearest-even
  return (unsigned short)(u >> 16);
}
__device__ __forceinline__ float bf2f(unsigned short s) {
  return __uint_as_float(((unsigned int)s) << 16);
}

// A fragment, 16-bit 16x32 (ISA 7.12.2): lane m=L%16, half=L/16.
// VGPR i<4 : K = half*8 + 2i ; VGPR i>=4 : K = 16 + half*8 + 2(i-4). Packed pairs -> b32 LDS reads.
__device__ __forceinline__ bf16x16 frag_a(const unsigned short* base, int stride, int lane, int kbase = 0) {
  const int m = lane & 15, half = lane >> 4;
  const unsigned short* row = base + m * stride + kbase + half * 8;
  FragU f;
#pragma unroll
  for (int i = 0; i < 4; ++i) {
    f.u[i]     = *(const unsigned int*)(row + 2 * i);
    f.u[i + 4] = *(const unsigned int*)(row + 16 + 2 * i);
  }
  return f.v;
}

// B fragment from a [N][K]-transposed LDS tile: lane n=nbase+L%16, half=L/16,
// VGPR j<4 : K = half*8 + 2j ; VGPR j>=4 : K = 16 + half*8 + 2(j-4) (matches frag_a).
__device__ __forceinline__ bf16x16 frag_b(const unsigned short* base, int stride, int lane,
                                          int nbase, int kbase) {
  const int n = nbase + (lane & 15), half = lane >> 4;
  const unsigned short* row = base + n * stride + kbase + half * 8;
  FragU f;
#pragma unroll
  for (int j = 0; j < 4; ++j) {
    f.u[j]     = *(const unsigned int*)(row + 2 * j);
    f.u[j + 4] = *(const unsigned int*)(row + 16 + 2 * j);
  }
  return f.v;
}

__device__ __forceinline__ f32x8 wmma_bf16(bf16x16 a, bf16x16 b, f32x8 c) {
  return __builtin_amdgcn_wmma_f32_16x16x32_bf16(false, a, false, b, (short)0, c, false, false);
}

#if USE_TDM
// Pack a 2-D TDM descriptor (D#) per CDNA5 ISA 8.3/8.4 and issue TENSOR_LOAD_TO_LDS.
// data_size=2B. Optional pad inserts pad_amount+1 DWORDs after every 2^(pad_interval+1)
// DWORDs, letting the TDM itself create bank-conflict-free LDS row strides.
__device__ __forceinline__ void tdm_load_2d(unsigned lds_off, const void* gptr,
                                            unsigned tile_d0, unsigned tile_d1,
                                            unsigned tensor_d0, unsigned tensor_d1,
                                            unsigned long long stride0,
                                            unsigned pad_en, unsigned pad_interval,
                                            unsigned pad_amount) {
  unsigned long long ga = (unsigned long long)(uintptr_t)gptr;
  u32x4 g0;
  g0[0] = 1u;                                               // count=1 (valid), user mode
  g0[1] = lds_off;                                          // lds_addr (bytes)
  g0[2] = (unsigned)(ga & 0xffffffffull);                   // global_addr[31:0]
  g0[3] = (unsigned)((ga >> 32) & 0x1ffffffull) | (2u << 30); // global_addr[56:32] | type=2
  i32x8 g1;
  g1[0] = (int)((1u << 16) | (pad_en << 20) | (pad_interval << 22) | (pad_amount << 25)); // data_size=2B
  g1[1] = (int)((tensor_d0 & 0xffffu) << 16);                              // tensor_dim0[15:0] @ bits79:48
  g1[2] = (int)(((tensor_d0 >> 16) & 0xffffu) | ((tensor_d1 & 0xffffu) << 16));
  g1[3] = (int)(((tensor_d1 >> 16) & 0xffffu) | ((tile_d0 & 0xffffu) << 16));
  g1[4] = (int)(tile_d1 & 0xffffu);                                        // tile_dim1, tile_dim2=0
  g1[5] = (int)(unsigned)(stride0 & 0xffffffffull);                        // tensor_dim0_stride
  g1[6] = (int)(unsigned)((stride0 >> 32) & 0xffffull);
  g1[7] = 0;
  i32x4 z4 = {0, 0, 0, 0};
#if __clang_major__ >= 23
  i32x8 z8 = {0, 0, 0, 0, 0, 0, 0, 0};
  __builtin_amdgcn_tensor_load_to_lds(g0, g1, z4, z4, z8, 0);
#else
  __builtin_amdgcn_tensor_load_to_lds(g0, g1, z4, z4, 0);
#endif
}
__device__ __forceinline__ unsigned lds_off_of(const void* p) {
  return (unsigned)(uintptr_t)p;     // LDS aperture: low 32 bits are the LDS offset
}
#endif

// ---- Kernel 1a: x [B,C,N] f32 -> xt [B*N, C] bf16 (LDS-tiled transpose) ----
__global__ void k_pack_x(const float* __restrict__ x, unsigned short* __restrict__ xt) {
  __shared__ float tile[32][33];
  const int b = blockIdx.z, n0 = blockIdx.x * 32, c0 = blockIdx.y * 32;
  const int tx = threadIdx.x, ty = threadIdx.y;    // 32 x 8
#pragma unroll
  for (int i = 0; i < 32; i += 8)
    tile[ty + i][tx] = x[((size_t)(b * CCH + c0 + ty + i)) * NN + n0 + tx];
  __syncthreads();
#pragma unroll
  for (int i = 0; i < 32; i += 8)
    xt[((size_t)(b * NN + n0 + ty + i)) * CCH + c0 + tx] = f2bf(tile[tx][ty + i]);
}

// ---- Kernel 1b: plain f32 -> bf16 convert ----
__global__ void k_cvt(const float* __restrict__ in, unsigned short* __restrict__ out, int n) {
  int i = blockIdx.x * blockDim.x + threadIdx.x;
  if (i < n) out[i] = f2bf(in[i]);
}

// ---- Kernel 2: Y[16384, 1152] = xt @ Wq (q|k|v slices), bf16 WMMA, f32 accum ----
// A tiles (128x32) are DMA'd by the Tensor Data Mover, double-buffered:
// issue tile k0+32 into the alternate buffer, then s_wait_tensorcnt(1) so the
// oldest (in-order) transfer is done while the next overlaps WMMA compute.
__global__ __launch_bounds__(256) void k_gemm_qkvv(const unsigned short* __restrict__ xt,
                                                   const unsigned short* __restrict__ wq,
                                                   unsigned short* __restrict__ Y) {
  __shared__ __align__(16) unsigned short ldsA[2][128 * 34]; // A tile [m][k], stride 34 (TDM pad)
  __shared__ __align__(16) unsigned short ldsBt[64 * 34];    // B tile transposed [n][k]
  const int tid = threadIdx.x, wave = tid >> 5, lane = tid & 31;
  const int row0  = blockIdx.y * 128;
  const int ycol0 = blockIdx.x * 64;
  const int wcol0 = (ycol0 < 768) ? ycol0 : ycol0 + 384;     // skip qkvv[2] slice

#if USE_TDM
  if (wave == 0)
    tdm_load_2d(lds_off_of(ldsA[0]), xt + (size_t)row0 * CCH,
                /*tile*/ 32, 128, /*tensor*/ CCH, BB * NN,
                /*stride0*/ (unsigned long long)CCH, /*pad*/ 1, 3, 0); // -> stride 34 elems
#endif

  f32x8 acc[4] = {};
  int buf = 0;
  for (int k0 = 0; k0 < CCH; k0 += 32, buf ^= 1) {
    __syncthreads();
#if !USE_TDM
    for (int e = tid; e < 128 * 32; e += 256) {
      int k = e & 31, m = e >> 5;
      ldsA[buf][m * 34 + k] = xt[(size_t)(row0 + m) * CCH + k0 + k];
    }
#endif
    for (int e = tid; e < 32 * 64; e += 256) {               // B transposed into LDS
      int n = e & 63, k = e >> 6;
      ldsBt[n * 34 + k] = wq[(size_t)(k0 + k) * WQC + wcol0 + n];
    }
    if (k0 + 32 < CCH)                                       // prefetch next B panel
      __builtin_prefetch(wq + (size_t)(k0 + 32) * WQC + wcol0, 0, 1);
#if USE_TDM
    if (wave == 0) {
      if (k0 + 32 < CCH) {                                   // double-buffer: issue next tile
        tdm_load_2d(lds_off_of(ldsA[buf ^ 1]), xt + (size_t)row0 * CCH + k0 + 32,
                    32, 128, CCH, BB * NN, (unsigned long long)CCH, 1, 3, 0);
        __builtin_amdgcn_s_wait_tensorcnt(1);                // oldest transfer complete
      } else {
        __builtin_amdgcn_s_wait_tensorcnt(0);
      }
    }
#endif
    __syncthreads();
    bf16x16 a = frag_a(ldsA[buf] + wave * 16 * 34, 34, lane);
#pragma unroll
    for (int j = 0; j < 4; ++j)
      acc[j] = wmma_bf16(a, frag_b(ldsBt, 34, lane, j * 16, 0), acc[j]);
  }
  const int half = lane >> 4, nn = lane & 15;
#pragma unroll
  for (int j = 0; j < 4; ++j)
#pragma unroll
    for (int r = 0; r < 8; ++r)
      Y[(size_t)(row0 + wave * 16 + r + 8 * half) * YCOL + ycol0 + j * 16 + nn] = f2bf(acc[j][r]);
}

// ---- Kernel 3: invnorm[b][c] = 1/max(||q[:,c]||, 1e-12), wave per column ----
__global__ void k_invnorm(const unsigned short* __restrict__ Y, float* __restrict__ invn) {
  const int gw = (int)((blockIdx.x * blockDim.x + threadIdx.x) >> 5);
  const int lane = threadIdx.x & 31;
  if (gw >= BB * CCH) return;
  const int b = gw / CCH, c = gw % CCH;
  const unsigned short* col = Y + (size_t)b * NN * YCOL + c;   // q slice at col offset 0
  float ss = 0.f;
  for (int n = lane; n < NN; n += 32) { float v = bf2f(col[(size_t)n * YCOL]); ss += v * v; }
#pragma unroll
  for (int off = 16; off >= 1; off >>= 1) ss += __shfl_xor(ss, off, 32);
  if (lane == 0) invn[gw] = 1.0f / fmaxf(sqrtf(ss), 1e-12f);
}

// ---- Kernel 4: kv[b][s][c][kk] = sum_n Y[b,n, off_s + c] * EF[n,kk]  (A staged transposed) ----
__global__ __launch_bounds__(256) void k_proj(const unsigned short* __restrict__ Y,
                                              const unsigned short* __restrict__ ef,
                                              unsigned short* __restrict__ kv) {
  __shared__ __align__(16) unsigned short ldsAT[128 * 34];   // [c][n]
  __shared__ __align__(16) unsigned short ldsBt[64 * 34];    // [kk][n]
  const int tid = threadIdx.x, wave = tid >> 5, lane = tid & 31;
  const int c0 = blockIdx.x * 128, b = blockIdx.y, s = blockIdx.z;
  const int yoff = (s == 0) ? 384 : 768;                     // k slice / v slice
  f32x8 acc[4] = {};
  for (int n0 = 0; n0 < NN; n0 += 32) {
    __syncthreads();
    for (int e = tid; e < 128 * 32; e += 256) {
      int c = e & 127, n = e >> 7;
      ldsAT[c * 34 + n] = Y[(size_t)(b * NN + n0 + n) * YCOL + yoff + c0 + c];
    }
    for (int e = tid; e < 64 * 32; e += 256) {
      int kk = e & 63, n = e >> 6;
      ldsBt[kk * 34 + n] = ef[(size_t)(n0 + n) * PKK + kk];
    }
    __syncthreads();
    bf16x16 a = frag_a(ldsAT + wave * 16 * 34, 34, lane);
#pragma unroll
    for (int j = 0; j < 4; ++j)
      acc[j] = wmma_bf16(a, frag_b(ldsBt, 34, lane, j * 16, 0), acc[j]);
  }
  const int half = lane >> 4, nn = lane & 15;
#pragma unroll
  for (int j = 0; j < 4; ++j)
#pragma unroll
    for (int r = 0; r < 8; ++r) {
      int c = c0 + wave * 16 + r + 8 * half;
      kv[(size_t)((b * 2 + s) * CCH + c) * PKK + j * 16 + nn] = f2bf(acc[j][r]);
    }
}

// ---- Kernel 5: fused attention per (b, h, 128-row block); in-register softmax ----
__global__ __launch_bounds__(256) void k_attn(const unsigned short* __restrict__ Y,
                                              const unsigned short* __restrict__ kv,
                                              const float* __restrict__ invn,
                                              const float* __restrict__ temp2,
                                              float* __restrict__ out) {
  __shared__ __align__(16) unsigned short ldsQ[8 * 16 * 66];  // per-wave qn / attn staging
  __shared__ __align__(16) unsigned short ldsKt[64 * 66];     // k_proj^T : [kk][dd(pad64)]
  __shared__ __align__(16) unsigned short ldsV[48 * 66];      // v_proj   : [dd][kk], stride 66
  const int tid = threadIdx.x, wave = tid >> 5, lane = tid & 31;
  const int n0 = blockIdx.x * 128, hh = blockIdx.y, b = blockIdx.z;
  const int coff = hh * DDIM;

#if USE_TDM
  // v_proj tile is a plain contiguous-row 48x64 tile: DMA it; pad_interval=4
  // (every 32 DWORDs = one 64-elem bf16 row) + 1 DWORD pad -> row stride 66 elems.
  if (wave == 0)
    tdm_load_2d(lds_off_of(ldsV), kv + (size_t)((b * 2 + 1) * CCH + coff) * PKK,
                /*tile*/ PKK, DDIM, /*tensor*/ PKK, BB * 2 * CCH,
                /*stride0*/ (unsigned long long)PKK, /*pad*/ 1, 4, 0);
#else
  for (int e = tid; e < 48 * 64; e += 256) {                  // stage V
    int kk = e & 63, dd = e >> 6;
    ldsV[dd * 66 + kk] = kv[(size_t)((b * 2 + 1) * CCH + coff + dd) * PKK + kk];
  }
#endif
  for (int e = tid; e < 64 * 64; e += 256) {                  // stage Kt (pad dd>=48 -> 0)
    int kk = e & 63, dd = e >> 6;
    unsigned short v = 0;
    if (dd < DDIM) v = kv[(size_t)((b * 2 + 0) * CCH + coff + dd) * PKK + kk];
    ldsKt[kk * 66 + dd] = v;
  }
  unsigned short* myQ = ldsQ + wave * 16 * 66;                // stage normalized q (bf16)
  for (int e = lane; e < 16 * 64; e += 32) {
    int dd = e & 63, m = e >> 6;
    unsigned short v = 0;
    if (dd < DDIM) {
      float q = bf2f(Y[(size_t)(b * NN + n0 + wave * 16 + m) * YCOL + coff + dd]);
      v = f2bf(q * invn[b * CCH + coff + dd]);
    }
    myQ[m * 66 + dd] = v;
  }
#if USE_TDM
  if (wave == 0) __builtin_amdgcn_s_wait_tensorcnt(0);
#endif
  __syncthreads();

  f32x8 acc[4] = {};                                          // attn = qn @ k_proj  [16 x 64]
#pragma unroll
  for (int ks = 0; ks < 64; ks += 32) {
    bf16x16 a = frag_a(myQ, 66, lane, ks);
#pragma unroll
    for (int j = 0; j < 4; ++j)
      acc[j] = wmma_bf16(a, frag_b(ldsKt, 66, lane, j * 16, ks), acc[j]);
  }
  const float t2 = temp2[hh];
  const int half = lane >> 4, nn = lane & 15;
  float inv_s[8];
#pragma unroll
  for (int r = 0; r < 8; ++r) {                               // softmax over 64 cols / row
    float m = -3.0e38f;
#pragma unroll
    for (int j = 0; j < 4; ++j) { acc[j][r] *= t2; m = fmaxf(m, acc[j][r]); }
#pragma unroll
    for (int off = 8; off >= 1; off >>= 1) m = fmaxf(m, __shfl_xor(m, off, 16));
    float s = 0.f;
#pragma unroll
    for (int j = 0; j < 4; ++j) { float e = __expf(acc[j][r] - m); acc[j][r] = e; s += e; }
#pragma unroll
    for (int off = 8; off >= 1; off >>= 1) s += __shfl_xor(s, off, 16);
    inv_s[r] = 1.0f / s;
  }
#pragma unroll
  for (int j = 0; j < 4; ++j)                                 // attn back to LDS as bf16 A-tile
#pragma unroll
    for (int r = 0; r < 8; ++r)
      myQ[(r + 8 * half) * 66 + j * 16 + nn] = f2bf(acc[j][r] * inv_s[r]);
  __syncthreads();

  f32x8 acc2[3] = {};                                         // x_sa = attn @ v_proj^T [16 x 48]
#pragma unroll
  for (int ks = 0; ks < 64; ks += 32) {
    bf16x16 a = frag_a(myQ, 66, lane, ks);
#pragma unroll
    for (int jd = 0; jd < 3; ++jd)
      acc2[jd] = wmma_bf16(a, frag_b(ldsV, 66, lane, jd * 16, ks), acc2[jd]);
  }
#pragma unroll
  for (int jd = 0; jd < 3; ++jd)                              // scatter with reference permutation
#pragma unroll
    for (int r = 0; r < 8; ++r) {
      int n  = n0 + wave * 16 + r + 8 * half;
      int dd = jd * 16 + nn;
      int i  = (dd * NH + hh) * NN + n;                       // flatten [d,h,N]
      int np = i / CCH, cp = i % CCH;                         // reinterpret as [N,C]
      out[((size_t)(b * CCH + cp)) * NN + np] = acc2[jd][r];
    }
}

extern "C" void kernel_launch(void* const* d_in, const int* in_sizes, int n_in,
                              void* d_out, int out_size, void* d_ws, size_t ws_size,
                              hipStream_t stream) {
  (void)in_sizes; (void)n_in; (void)out_size; (void)ws_size;
  const float* x     = (const float*)d_in[0];
  const float* Wq    = (const float*)d_in[1];
  const float* EF    = (const float*)d_in[2];
  const float* temp2 = (const float*)d_in[3];
  char* ws = (char*)d_ws;
  unsigned short* xt   = (unsigned short*)ws; ws += (size_t)16384 * CCH * 2;   // 12.6 MB
  unsigned short* wqb  = (unsigned short*)ws; ws += (size_t)CCH * WQC * 2;     //  1.2 MB
  unsigned short* efb  = (unsigned short*)ws; ws += (size_t)NN * PKK * 2;      //  0.5 MB
  unsigned short* Yb   = (unsigned short*)ws; ws += (size_t)16384 * YCOL * 2;  // 37.7 MB
  float*          invn = (float*)ws;          ws += (size_t)BB * CCH * 4;
  unsigned short* kvp  = (unsigned short*)ws;                                   // 0.4 MB
  float* out = (float*)d_out;

  k_pack_x  <<<dim3(NN / 32, CCH / 32, BB), dim3(32, 8), 0, stream>>>(x, xt);
  k_cvt     <<<(CCH * WQC + 255) / 256, 256, 0, stream>>>(Wq, wqb, CCH * WQC);
  k_cvt     <<<(NN * PKK + 255) / 256, 256, 0, stream>>>(EF, efb, NN * PKK);
  k_gemm_qkvv<<<dim3(YCOL / 64, 16384 / 128), 256, 0, stream>>>(xt, wqb, Yb);
  k_invnorm <<<(BB * CCH * 32 + 255) / 256, 256, 0, stream>>>(Yb, invn);
  k_proj    <<<dim3(CCH / 128, BB, 2), 256, 0, stream>>>(Yb, efb, kvp);
  k_attn    <<<dim3(NN / 128, NH, BB), 256, 0, stream>>>(Yb, kvp, invn, temp2, out);
}
